// SelfAttention_81003083203757
// MI455X (gfx1250) — matrix operand order repaired and checked by reading, not compiled, not run on hardware
//
#include <hip/hip_runtime.h>

// ---------- types ----------
typedef __attribute__((ext_vector_type(16))) __bf16 bf16x16;
typedef __attribute__((ext_vector_type(8)))  float  f32x8;
typedef __attribute__((ext_vector_type(4)))  unsigned int u32x4;
typedef __attribute__((ext_vector_type(4)))  int i32x4;
typedef __attribute__((ext_vector_type(8)))  int i32x8;

union Frag { bf16x16 v; u32x4 q[2]; };
union H8   { u32x4 q; unsigned short h[8]; };

__device__ __forceinline__ unsigned short f2bf(float f) {
  unsigned int u = __float_as_uint(f);
  u += 0x7FFFu + ((u >> 16) & 1u);           // round-to-nearest-even
  return (unsigned short)(u >> 16);
}
__device__ __forceinline__ float bf2f(unsigned short h) {
  return __uint_as_float((unsigned int)h << 16);
}

__device__ __forceinline__ f32x8 wmma_bf16(const Frag& a, const Frag& b, f32x8 c) {
  return __builtin_amdgcn_wmma_f32_16x16x32_bf16(false, a.v, false, b.v,
                                                 (short)0, c, false, false);
}

// ---------- CDNA5 async global->LDS staging (guarded; falls back to ds_store) ----
#if defined(__gfx1250__) && __has_builtin(__builtin_amdgcn_global_load_async_to_lds_b128)
#define USE_ASYNC_LDS 1
#else
#define USE_ASYNC_LDS 0
#endif

// ---------- CDNA5 Tensor Data Mover (guarded; falls back to async per-lane) ----
#if defined(__gfx1250__) && __has_builtin(__builtin_amdgcn_tensor_load_to_lds)
#define USE_TDM 1
#else
#define USE_TDM 0
#endif

typedef __attribute__((address_space(1))) i32x4 gas_i32x4;
typedef __attribute__((address_space(3))) i32x4 las_i32x4;
typedef __attribute__((address_space(3))) unsigned short las_u16;

__device__ __forceinline__ void stage16(unsigned short* dst_lds,
                                        const unsigned short* src_gbl) {
#if USE_ASYNC_LDS
  __builtin_amdgcn_global_load_async_to_lds_b128(
      (gas_i32x4*)(i32x4*)src_gbl,
      (las_i32x4*)(i32x4*)dst_lds,
      0, 0);
#else
  *(u32x4*)dst_lds = *(const u32x4*)src_gbl;
#endif
}

#if USE_TDM
__device__ __forceinline__ unsigned lds_off(unsigned short* p) {
  return (unsigned)(size_t)(las_u16*)p;
}
// 2D TDM tile load: rows x width bf16, global row stride rstride (elements),
// LDS rows padded by 16B after every row (interval_code selects row byte size).
__device__ __forceinline__ void tdm_load(unsigned laddr, const unsigned short* src,
                                         int width, int rows, int rstride,
                                         int interval_code) {
  unsigned long long g = (unsigned long long)(size_t)src;
  // D# group0: count=1, lds_addr, global_addr[56:0], type=2
  u32x4 g0 = { 1u, laddr, (unsigned)g,
               (unsigned)((g >> 32) & 0x01FFFFFFu) | (2u << 30) };
  // D# group1: data_size=2B, pad_enable, pad_interval, pad_amount=4 dwords(16B),
  //            tensor_dim0=width, tensor_dim1=rows, tile_dim0=width, tile_dim1=rows,
  //            tensor_dim0_stride=rstride
  i32x8 g1 = { (int)((1u << 16) | (1u << 20) |
                     ((unsigned)interval_code << 22) | (3u << 25)),
               width << 16, rows << 16, width << 16,
               rows, rstride, 0, 0 };
  i32x4 z4 = { 0, 0, 0, 0 };
#if __clang_major__ >= 23
  i32x8 z8 = { 0, 0, 0, 0, 0, 0, 0, 0 };
  __builtin_amdgcn_tensor_load_to_lds(g0, g1, z4, z4, z8, 0);
#else
  __builtin_amdgcn_tensor_load_to_lds(g0, g1, z4, z4, 0);
#endif
}
#endif

__device__ __forceinline__ void stage_wait_all() {
#if USE_TDM
#if __has_builtin(__builtin_amdgcn_s_wait_tensorcnt)
  __builtin_amdgcn_s_wait_tensorcnt(0);
#else
  asm volatile("s_wait_tensorcnt 0x0" ::: "memory");
#endif
#endif
#if USE_ASYNC_LDS
#if __has_builtin(__builtin_amdgcn_s_wait_asynccnt)
  __builtin_amdgcn_s_wait_asynccnt(0);
#else
  asm volatile("s_wait_asynccnt 0x0" ::: "memory");
#endif
#endif
}

// ---------- elementwise fp32 -> bf16 ----------
__global__ void cvt_kernel(const float* __restrict__ in,
                           unsigned short* __restrict__ out, int n) {
  int i = blockIdx.x * blockDim.x + threadIdx.x;
  if (i < n) out[i] = f2bf(in[i]);
}

// ---------- row sum-of-squares over 64 bf16 ----------
__global__ void rowsq_kernel(const unsigned short* __restrict__ src,
                             float* __restrict__ dst, int rows) {
  int r = blockIdx.x * blockDim.x + threadIdx.x;
  if (r >= rows) return;
  const unsigned short* p = src + (size_t)r * 64;
  float s = 0.f;
  #pragma unroll
  for (int i = 0; i < 8; ++i) {
    H8 h; h.q = *(const u32x4*)(p + i * 8);
    #pragma unroll
    for (int j = 0; j < 8; ++j) { float f = bf2f(h.h[j]); s += f * f; }
  }
  dst[r] = s;
}

// ---------- LayerNorm over 512 cols, fp32 in -> bf16 out ----------
__global__ void __launch_bounds__(256)
ln_kernel(const float* __restrict__ xin, const float* __restrict__ g,
          const float* __restrict__ bta, unsigned short* __restrict__ out) {
  __shared__ float red[256];
  int row = blockIdx.x, tid = threadIdx.x;
  const float* x = xin + (size_t)row * 512;
  float a = x[tid], b = x[tid + 256];
  red[tid] = a + b;
  __syncthreads();
  for (int s = 128; s > 0; s >>= 1) {
    if (tid < s) red[tid] += red[tid + s];
    __syncthreads();
  }
  float mu = red[0] * (1.0f / 512.0f);
  __syncthreads();
  float da = a - mu, db = b - mu;
  red[tid] = da * da + db * db;
  __syncthreads();
  for (int s = 128; s > 0; s >>= 1) {
    if (tid < s) red[tid] += red[tid + s];
    __syncthreads();
  }
  float rs = rsqrtf(red[0] * (1.0f / 512.0f) + 1e-4f);
  out[(size_t)row * 512 + tid]       = f2bf(g[tid] * da * rs + bta[tid]);
  out[(size_t)row * 512 + tid + 256] = f2bf(g[tid + 256] * db * rs + bta[tid + 256]);
}

// ---------- tiled WMMA GEMM: C[M,N] = A[M,K] * B[K,N], bf16 in, mode epilogue ----
// Double-buffered + software pipelined: A tile via TDM/async-LDS, B tile via
// registers (global load early, ds_store after compute).
enum { MODE_Q = 0, MODE_CONV = 1, MODE_KV = 2, MODE_PROJ = 3 };

template <int MODE>
__global__ void __launch_bounds__(256)
gemm_kernel(const unsigned short* __restrict__ A,
            const unsigned short* __restrict__ B,
            void* __restrict__ C0, void* __restrict__ C1,
            int M, int N, int K) {
  __shared__ __attribute__((aligned(16))) unsigned short As[2][128 * 40];
  __shared__ __attribute__((aligned(16))) unsigned short Bs[2][128 * 40]; // B^T
  const int tid  = threadIdx.x;
  const int lane = tid & 31;
  const int wid  = tid >> 5;
  const int m16  = lane & 15, hi = lane >> 4;
  const int wm = (wid & 3) * 32;   // wave row block
  const int wn = (wid >> 2) * 64;  // wave col block
  const int m0 = blockIdx.y * 128, n0 = blockIdx.x * 128;
  const int NT = K >> 5;

  const int kk2 = (tid >> 4) * 2;
  const int nb  = (tid & 15) * 8;
  H8 rb0, rb1;

  auto stageA = [&](int t, int p) {
    int k0 = t * 32;
    if constexpr (MODE == MODE_CONV) {
      // non-affine row map: per-lane async copies
      #pragma unroll
      for (int i = 0; i < 2; ++i) {
        int idx = tid + i * 256;
        int row = idx >> 2, seg = (idx & 3) * 8;
        int rg = m0 + row;
        int bb = rg >> 10, pq = rg & 1023;
        int y = pq >> 5, xq = pq & 31;
        int dy = k0 >> 10, dx = (k0 >> 9) & 1, ci = k0 & 511;
        const unsigned short* src =
            A + ((size_t)((bb * 64 + 2 * y + dy) * 64 + 2 * xq + dx)) * 512 + ci + seg;
        stage16(&As[p][row * 40 + seg], src);
      }
    } else {
#if USE_TDM
      // one TDM descriptor per wave: 16 rows x 32 bf16, row 64B -> pad to 80B
      tdm_load(lds_off(&As[p][wid * 16 * 40]),
               A + (size_t)(m0 + wid * 16) * K + k0, 32, 16, K, 3);
#else
      #pragma unroll
      for (int i = 0; i < 2; ++i) {
        int idx = tid + i * 256;
        int row = idx >> 2, seg = (idx & 3) * 8;
        stage16(&As[p][row * 40 + seg], A + (size_t)(m0 + row) * K + k0 + seg);
      }
#endif
    }
  };
  auto loadB = [&](int t) {
    const unsigned short* s = B + (size_t)(t * 32 + kk2) * N + n0 + nb;
    rb0.q = *(const u32x4*)s;
    rb1.q = *(const u32x4*)(s + N);
  };
  auto storeB = [&](int p) {
    #pragma unroll
    for (int i = 0; i < 8; ++i) {
      unsigned int pk = (unsigned int)rb0.h[i] | ((unsigned int)rb1.h[i] << 16);
      *(unsigned int*)&Bs[p][(nb + i) * 40 + kk2] = pk;
    }
  };

  f32x8 acc[2][4] = {};

  // prologue: tile 0 into buffer 0
  stageA(0, 0);
  loadB(0);
  storeB(0);

  for (int t = 0; t < NT; ++t) {
    const int p = t & 1;
    stage_wait_all();
    __syncthreads();
    if (t + 1 < NT) {           // kick next tile while computing this one
      stageA(t + 1, p ^ 1);
      loadB(t + 1);
    }
    // --- fragments + WMMA from buffer p ---
    const unsigned short* Ap = &As[p][0];
    const unsigned short* Bp = &Bs[p][0];
    Frag af[2], bfr[4];
    #pragma unroll
    for (int i = 0; i < 2; ++i) {
      const unsigned short* pp = &Ap[(wm + i * 16 + m16) * 40];
      af[i].q[0] = *(const u32x4*)(pp + hi * 16);      // K0-7 / K16-23
      af[i].q[1] = *(const u32x4*)(pp + hi * 16 + 8); // K8-15 / K24-31
    }
    #pragma unroll
    for (int j = 0; j < 4; ++j) {
      const unsigned short* pp = &Bp[(wn + j * 16 + m16) * 40 + hi * 16];
      bfr[j].q[0] = *(const u32x4*)pp;                 // K0-15 / K16-31
      bfr[j].q[1] = *(const u32x4*)(pp + 8);
    }
    #pragma unroll
    for (int i = 0; i < 2; ++i)
      #pragma unroll
      for (int j = 0; j < 4; ++j)
        acc[i][j] = wmma_bf16(af[i], bfr[j], acc[i][j]);
    if (t + 1 < NT) storeB(p ^ 1);  // deferred ds_store (hides load latency)
  }

  // --- epilogue ---
  #pragma unroll
  for (int i = 0; i < 2; ++i) {
    #pragma unroll
    for (int j = 0; j < 4; ++j) {
      int col   = n0 + wn + j * 16 + m16;
      int rbase = m0 + wm + i * 16 + hi * 8;
      if constexpr (MODE == MODE_CONV || MODE == MODE_PROJ) {
        float* Cf = (float*)C0;
        #pragma unroll
        for (int r = 0; r < 8; ++r)
          Cf[(size_t)(rbase + r) * N + col] = acc[i][j][r];
      } else if constexpr (MODE == MODE_Q) {
        unsigned short* Cq = (unsigned short*)C0;
        int h = col >> 6, d = col & 63;
        #pragma unroll
        for (int r = 0; r < 8; ++r) {
          int rg = rbase + r;
          int b = rg >> 12, nloc = rg & 4095;
          Cq[((size_t)(b * 8 + h) * 4096 + nloc) * 64 + d] = f2bf(acc[i][j][r]);
        }
      } else { // MODE_KV
        int b = rbase >> 10, dn0 = rbase & 1023;
        if (col < 512) {
          unsigned short* Ck = (unsigned short*)C0;
          int h = col >> 6, d = col & 63;
          #pragma unroll
          for (int r = 0; r < 8; ++r)
            Ck[((size_t)(b * 8 + h) * 1024 + dn0 + r) * 64 + d] = f2bf(acc[i][j][r]);
        } else {
          unsigned short* Cv = (unsigned short*)C1;
          int c2 = col - 512;
          int h = c2 >> 6, d = c2 & 63;
          H8 pk;
          #pragma unroll
          for (int r = 0; r < 8; ++r) pk.h[r] = f2bf(acc[i][j][r]);
          *(u32x4*)&Cv[((size_t)(b * 8 + h) * 64 + d) * 1024 + dn0] = pk.q;
        }
      }
    }
  }
}

// ---------- fused flash attention with L2-distance scores ----------
// grid: 2048 blocks (32 bh * 64 q-blocks), 128 threads (4 waves, 16 queries each)
// Double-buffered K/V tiles staged by TDM (or per-lane async fallback).
__global__ void __launch_bounds__(128)
flash_kernel(const unsigned short* __restrict__ qb,   // [bh][4096][64]
             const unsigned short* __restrict__ kg,   // [bh][1024][64]
             const unsigned short* __restrict__ vt,   // [bh][64][1024]
             const float* __restrict__ qq,            // [bh*4096] ||q||^2
             const float* __restrict__ kk,            // [bh*1024] ||k||^2
             unsigned short* __restrict__ attn)       // [b*4096][512] bf16
{
  __shared__ __attribute__((aligned(16))) unsigned short Ks[2][32 * 72];
  __shared__ __attribute__((aligned(16))) unsigned short Vs[2][64 * 40];
  __shared__ __attribute__((aligned(16))) unsigned short Ps[4 * 16 * 40];

  const int tid = threadIdx.x;
  const int lane = tid & 31, wid = tid >> 5;
  const int m16 = lane & 15, hi = lane >> 4;
  const int bh = blockIdx.x >> 6;
  const int q0 = (blockIdx.x & 63) * 64;
  const int b = bh >> 3, h = bh & 7;
  const float scale = 0.125f; // 64^-0.5

  // q fragments (16 x 64 per wave, split into two K=32 A-frags)
  const int qrow = q0 + wid * 16 + m16;
  const unsigned short* qp = qb + ((size_t)bh * 4096 + qrow) * 64;
  Frag qf[2];
  qf[0].q[0] = *(const u32x4*)(qp + hi * 16);
  qf[0].q[1] = *(const u32x4*)(qp + hi * 16 + 8);
  qf[1].q[0] = *(const u32x4*)(qp + 32 + hi * 16);
  qf[1].q[1] = *(const u32x4*)(qp + 32 + hi * 16 + 8);

  // per-(row, half) broadcast of ||q||^2 to match C-tile layout
  float qqv = qq[(size_t)bh * 4096 + qrow];
  float rqq[8];
  #pragma unroll
  for (int r = 0; r < 8; ++r) rqq[r] = __shfl(qqv, r + hi * 8, 32);

  float mrun[8], lrun[8];
  #pragma unroll
  for (int r = 0; r < 8; ++r) { mrun[r] = -3.0e38f; lrun[r] = 0.f; }
  f32x8 o[4] = {};

  unsigned short* psw = Ps + wid * 16 * 40;

  // per-thread fallback staging addresses
  const int kr = tid >> 2, ksg = (tid & 3) * 16;       // K tile: 32 rows x 64
  const int vd = tid >> 1, vsg = (tid & 1) * 16;       // V^T tile: 64 rows x 32
  const unsigned short* ksrc = kg + ((size_t)bh * 1024 + kr) * 64 + ksg;
  const unsigned short* vsrc = vt + ((size_t)bh * 64 + vd) * 1024 + vsg;

  auto stageKV = [&](int t, int p) {
    int key0 = t * 32;
#if USE_TDM
    // K: per wave 8 rows x 64 bf16 (row 128B -> pad to 144B)
    tdm_load(lds_off(&Ks[p][wid * 8 * 72]),
             kg + ((size_t)bh * 1024 + key0 + wid * 8) * 64, 64, 8, 64, 4);
    // V^T: per wave 16 rows x 32 bf16, row stride 1024 (row 64B -> pad to 80B)
    tdm_load(lds_off(&Vs[p][wid * 16 * 40]),
             vt + ((size_t)bh * 64 + wid * 16) * 1024 + key0, 32, 16, 1024, 3);
#else
    const unsigned short* s = ksrc + (size_t)key0 * 64;
    stage16(&Ks[p][kr * 72 + ksg],     s);
    stage16(&Ks[p][kr * 72 + ksg + 8], s + 8);
    const unsigned short* sv = vsrc + key0;
    stage16(&Vs[p][vd * 40 + vsg],     sv);
    stage16(&Vs[p][vd * 40 + vsg + 8], sv + 8);
#endif
  };

  stageKV(0, 0);                          // prologue

  for (int t = 0; t < 32; ++t) {
    const int p = t & 1;
    const int key0 = t * 32;
    stage_wait_all();
    __syncthreads();
    if (t + 1 < 32) {
      stageKV(t + 1, p ^ 1);              // DMA next tile under compute
      if (t + 2 < 32) {                   // warm L2 two tiles ahead
        __builtin_prefetch(ksrc + (size_t)(key0 + 64) * 64, 0, 1);
        __builtin_prefetch(vsrc + key0 + 64, 0, 1);
      }
    }
    const unsigned short* Kp = &Ks[p][0];
    const unsigned short* Vp = &Vs[p][0];

    float kk0 = kk[(size_t)bh * 1024 + key0 + m16];
    float kk1 = kk[(size_t)bh * 1024 + key0 + 16 + m16];

    // S = Q . K^T  (two 16x16 tiles: keys 0-15 and 16-31)
    Frag kf;
    f32x8 s0 = {}, s1 = {};
    kf.q[0] = *(const u32x4*)&Kp[m16 * 72 + hi * 16];
    kf.q[1] = *(const u32x4*)&Kp[m16 * 72 + hi * 16 + 8];
    s0 = wmma_bf16(qf[0], kf, s0);
    kf.q[0] = *(const u32x4*)&Kp[m16 * 72 + 32 + hi * 16];
    kf.q[1] = *(const u32x4*)&Kp[m16 * 72 + 32 + hi * 16 + 8];
    s0 = wmma_bf16(qf[1], kf, s0);
    kf.q[0] = *(const u32x4*)&Kp[(16 + m16) * 72 + hi * 16];
    kf.q[1] = *(const u32x4*)&Kp[(16 + m16) * 72 + hi * 16 + 8];
    s1 = wmma_bf16(qf[0], kf, s1);
    kf.q[0] = *(const u32x4*)&Kp[(16 + m16) * 72 + 32 + hi * 16];
    kf.q[1] = *(const u32x4*)&Kp[(16 + m16) * 72 + 32 + hi * 16 + 8];
    s1 = wmma_bf16(qf[1], kf, s1);

    // distance score + online softmax
    float v0[8], v1[8], p0[8], p1[8];
    #pragma unroll
    for (int r = 0; r < 8; ++r) {
      v0[r] = sqrtf(fmaxf(rqq[r] + kk0 - 2.f * s0[r], 1e-7f)) * scale;
      v1[r] = sqrtf(fmaxf(rqq[r] + kk1 - 2.f * s1[r], 1e-7f)) * scale;
    }
    #pragma unroll
    for (int r = 0; r < 8; ++r) {
      float mx = fmaxf(v0[r], v1[r]);
      #pragma unroll
      for (int msk = 8; msk >= 1; msk >>= 1)
        mx = fmaxf(mx, __shfl_xor(mx, msk, 32));
      float mnew  = fmaxf(mrun[r], mx);
      float alpha = __expf(mrun[r] - mnew);
      mrun[r] = mnew;
      p0[r] = __expf(v0[r] - mnew);
      p1[r] = __expf(v1[r] - mnew);
      float rs = p0[r] + p1[r];
      #pragma unroll
      for (int msk = 8; msk >= 1; msk >>= 1)
        rs += __shfl_xor(rs, msk, 32);
      lrun[r] = lrun[r] * alpha + rs;
      #pragma unroll
      for (int g = 0; g < 4; ++g) o[g][r] *= alpha;
    }

    // P (C-layout) -> LDS -> A-fragment (wave-local transpose)
    #pragma unroll
    for (int r = 0; r < 8; ++r) {
      psw[(r + hi * 8) * 40 + m16]      = f2bf(p0[r]);
      psw[(r + hi * 8) * 40 + 16 + m16] = f2bf(p1[r]);
    }
    Frag pf;
    pf.q[0] = *(const u32x4*)&psw[m16 * 40 + hi * 16];
    pf.q[1] = *(const u32x4*)&psw[m16 * 40 + hi * 16 + 8];

    // O += P . V  (4 d-groups of 16)
    #pragma unroll
    for (int g = 0; g < 4; ++g) {
      Frag vf;
      vf.q[0] = *(const u32x4*)&Vp[(g * 16 + m16) * 40 + hi * 16];
      vf.q[1] = *(const u32x4*)&Vp[(g * 16 + m16) * 40 + hi * 16 + 8];
      o[g] = wmma_bf16(pf, vf, o[g]);
    }
  }

  // finalize: O /= l, store to attn_out[b, n, h*64+d] bf16
  float inv[8];
  #pragma unroll
  for (int r = 0; r < 8; ++r) inv[r] = 1.0f / lrun[r];
  #pragma unroll
  for (int g = 0; g < 4; ++g) {
    int col = h * 64 + g * 16 + m16;
    #pragma unroll
    for (int r = 0; r < 8; ++r) {
      int rowg = b * 4096 + q0 + wid * 16 + hi * 8 + r;
      attn[(size_t)rowg * 512 + col] = f2bf(o[g][r] * inv[r]);
    }
  }
}

// ---------- host launcher ----------
extern "C" void kernel_launch(void* const* d_in, const int* in_sizes, int n_in,
                              void* d_out, int out_size, void* d_ws, size_t ws_size,
                              hipStream_t stream) {
  (void)in_sizes; (void)n_in; (void)out_size; (void)ws_size;
  const float* x     = (const float*)d_in[0];
  const float* Wq    = (const float*)d_in[1];
  const float* Wkv   = (const float*)d_in[2];
  const float* convw = (const float*)d_in[3];
  const float* gamma = (const float*)d_in[4];
  const float* beta  = (const float*)d_in[5];
  const float* Wproj = (const float*)d_in[6];

  char* ws = (char*)d_ws;
  size_t off = 0;
  auto alloc = [&](size_t bytes) {
    void* p = ws + off;
    off += bytes;
    off = (off + 255) & ~(size_t)255;
    return p;
  };
  unsigned short* xb   = (unsigned short*)alloc(16384ull * 512 * 2);
  unsigned short* wqb  = (unsigned short*)alloc(512ull * 512 * 2);
  unsigned short* wkvb = (unsigned short*)alloc(512ull * 1024 * 2);
  unsigned short* wcb  = (unsigned short*)alloc(2048ull * 512 * 2);
  unsigned short* wpb  = (unsigned short*)alloc(512ull * 512 * 2);
  unsigned short* qbuf = (unsigned short*)alloc(32ull * 4096 * 64 * 2);
  float*          cvf  = (float*)alloc(4096ull * 512 * 4);
  unsigned short* xk   = (unsigned short*)alloc(4096ull * 512 * 2);
  unsigned short* kg   = (unsigned short*)alloc(32ull * 1024 * 64 * 2);
  unsigned short* vtg  = (unsigned short*)alloc(32ull * 1024 * 64 * 2);
  float*          qqv  = (float*)alloc(131072ull * 4);
  float*          kkv  = (float*)alloc(32768ull * 4);
  unsigned short* attn = (unsigned short*)alloc(16384ull * 512 * 2);

  // fp32 -> bf16 conversions
  cvt_kernel<<<4194304 / 256, 256, 0, stream>>>(x, xb, 4194304);
  cvt_kernel<<<262144 / 256, 256, 0, stream>>>(Wq, wqb, 262144);
  cvt_kernel<<<524288 / 256, 256, 0, stream>>>(Wkv, wkvb, 524288);
  cvt_kernel<<<1048576 / 256, 256, 0, stream>>>(convw, wcb, 1048576);
  cvt_kernel<<<262144 / 256, 256, 0, stream>>>(Wproj, wpb, 262144);

  // q = x @ Wq  -> head-split layout
  gemm_kernel<MODE_Q><<<dim3(4, 128), 256, 0, stream>>>(xb, wqb, qbuf, nullptr, 16384, 512, 512);
  // patch-conv as GEMM -> fp32
  gemm_kernel<MODE_CONV><<<dim3(4, 32), 256, 0, stream>>>(xb, wcb, cvf, nullptr, 4096, 512, 2048);
  // LayerNorm -> bf16
  ln_kernel<<<4096, 256, 0, stream>>>(cvf, gamma, beta, xk);
  // kv = ln @ Wkv -> k row-major, v transposed
  gemm_kernel<MODE_KV><<<dim3(8, 32), 256, 0, stream>>>(xk, wkvb, kg, vtg, 4096, 1024, 512);
  // squared norms
  rowsq_kernel<<<131072 / 256, 256, 0, stream>>>(qbuf, qqv, 131072);
  rowsq_kernel<<<32768 / 256, 256, 0, stream>>>(kg, kkv, 32768);
  // fused L2-distance flash attention
  flash_kernel<<<2048, 128, 0, stream>>>(qbuf, kg, vtg, qqv, kkv, attn);
  // out = attn_out @ Wproj -> fp32 d_out
  gemm_kernel<MODE_PROJ><<<dim3(4, 128), 256, 0, stream>>>(attn, wpb, (float*)d_out, nullptr, 16384, 512, 512);
}
